// gwnet_12335146074621
// MI455X (gfx1250) — matrix operand order repaired and checked by reading, not compiled, not run on hardware
//
#include <hip/hip_runtime.h>
#include <hip/hip_bf16.h>

// Graph WaveNet forward for MI455X (gfx1250, wave32).
// - All h-intermediates bf16, node dim padded 500->512, T stride fixed 256.
// - Graph conv (dominant ~0.8 TFLOP) via v_wmma_f32_16x16x32_bf16, 16x64 strip per wave.
// - gcn channel-mix + residual + BN fused into ONE WMMA kernel (5 chained WMMAs per tile).
// - Dilated gated conv as WMMA too: K=64 (2 taps x 32ch), tap1 = +d column offset;
//   tanh*sigmoid in the epilogue.
// - Skip path algebraically reduced to last-time-column taps (final T == 1).
// Workspace: 2*512KB (supports) + 7*64MB (bf16 h bufs) + ~140KB (Wb/Wc) + 4MB (skip) ~= 475 MB.

typedef unsigned short u16;
typedef unsigned int   u32;
typedef __attribute__((ext_vector_type(16))) __bf16 v16bf;
typedef __attribute__((ext_vector_type(8)))  float  v8f;

#define B_  8
#define CH  32
#define NN  500
#define NP  512
#define TSZ 256
#define SC  256
#define EC  512
#define OD  12
#define LL  8
#define BN_SCALE 0.9999950000374997f   // 1/sqrt(1 + 1e-5)

__device__ __forceinline__ u16 f2bf(float f) {
  union { float f; u32 u; } v; v.f = f;
  u32 r = v.u + 0x7FFFu + ((v.u >> 16) & 1u);   // round-to-nearest-even
  return (u16)(r >> 16);
}
__device__ __forceinline__ float bf2f(u16 h) {
  union { u32 u; float f; } v; v.u = ((u32)h) << 16; return v.f;
}
__device__ __forceinline__ size_t hidx(int b, int c, int n, int t) {
  return (((size_t)(b * CH + c) * NP) + (size_t)n) * TSZ + (size_t)t;
}

// ---------------- adaptive adjacency: softmax(relu(E1@E2), axis=1), store transposed bf16 ----
__global__ void adp_softmax_kernel(const float* __restrict__ nv1, const float* __restrict__ nv2,
                                   u16* __restrict__ adpT) {
  __shared__ float z[NN];
  __shared__ float red[256];
  int r = blockIdx.x, tid = threadIdx.x;
  float e1[10];
#pragma unroll
  for (int k = 0; k < 10; ++k) e1[k] = nv1[r * 10 + k];
  for (int w = tid; w < NN; w += 256) {
    float s = 0.f;
#pragma unroll
    for (int k = 0; k < 10; ++k) s += e1[k] * nv2[k * NN + w];
    z[w] = fmaxf(s, 0.f);
  }
  __syncthreads();
  float m = -1e30f;
  for (int w = tid; w < NN; w += 256) m = fmaxf(m, z[w]);
  red[tid] = m; __syncthreads();
  for (int s = 128; s > 0; s >>= 1) { if (tid < s) red[tid] = fmaxf(red[tid], red[tid + s]); __syncthreads(); }
  float mx = red[0]; __syncthreads();
  float sum = 0.f;
  for (int w = tid; w < NN; w += 256) { float e = __expf(z[w] - mx); z[w] = e; sum += e; }
  red[tid] = sum; __syncthreads();
  for (int s = 128; s > 0; s >>= 1) { if (tid < s) red[tid] += red[tid + s]; __syncthreads(); }
  float inv = 1.f / red[0];
  __syncthreads();
  for (int w = tid; w < NN; w += 256) adpT[(size_t)w * NP + r] = f2bf(z[w] * inv);
}

// ---------------- A0 -> transposed, zero-padded bf16; also zero adpT pad region ---------------
__global__ void transpose_pad_kernel(const float* __restrict__ A0, u16* __restrict__ AT,
                                     u16* __restrict__ adpT) {
  int i = blockIdx.x * 256 + threadIdx.x;      // covers 512*512
  int w = i >> 9, v = i & 511;
  bool in = (w < NN) && (v < NN);
  AT[i] = in ? f2bf(A0[(size_t)v * NN + w]) : (u16)0;
  if (!in) adpT[i] = 0;
}

// ---------------- gcn weights [L,32,160] f32 -> Wb [L][k][o][c] bf16 --------------------------
__global__ void wb_convert_kernel(const float* __restrict__ gcn_w, u16* __restrict__ Wb) {
  int i = blockIdx.x * 256 + threadIdx.x;      // (LL-1)*5*32*32 = 35840
  if (i >= (LL - 1) * 5 * CH * CH) return;
  int c = i & 31; int r = i >> 5;
  int o = r & 31; r >>= 5;
  int k = r % 5;  int L = r / 5;
  Wb[i] = f2bf(gcn_w[L * 5120 + o * 160 + k * 32 + c]);
}

// ---------------- filt/gate weights [L,32,32,2] f32 -> Wc [L][4][o][c] bf16 -------------------
// sidx: 0 = filt tap0, 1 = filt tap1, 2 = gate tap0, 3 = gate tap1
__global__ void wc_convert_kernel(const float* __restrict__ filt_w,
                                  const float* __restrict__ gate_w, u16* __restrict__ Wc) {
  int i = blockIdx.x * 256 + threadIdx.x;      // LL*4*32*32 = 32768
  if (i >= LL * 4 * CH * CH) return;
  int c = i & 31; int r = i >> 5;
  int o = r & 31; r >>= 5;
  int s = r & 3;  int L = r >> 2;
  int tap = s & 1;
  const float* src = (s >= 2) ? gate_w : filt_w;
  Wc[i] = f2bf(src[L * 2048 + (o * 32 + c) * 2 + tap]);
}

// ---------------- start 1x1 conv: [B,2,N,T] -> bf16 [B,32,Npad,T] -----------------------------
__global__ void start_conv_kernel(const float* __restrict__ x, const float* __restrict__ sw,
                                  const float* __restrict__ sb, u16* __restrict__ h) {
  long i = (long)blockIdx.x * 256 + threadIdx.x;   // B*CH*NN*TSZ
  int t = (int)(i & 255); long r = i >> 8;
  int n = (int)(r % NN); r /= NN;
  int o = (int)(r & 31); int b = (int)(r >> 5);
  float v = sb[o]
          + sw[o * 2 + 0] * x[((size_t)(b * 2 + 0) * NN + n) * TSZ + t]
          + sw[o * 2 + 1] * x[((size_t)(b * 2 + 1) * NN + n) * TSZ + t];
  h[hidx(b, o, n, t)] = f2bf(v);
}

// ---------------- dilated gated conv via WMMA: hg = tanh(Wf*[h;h+d]) * sigm(Wg*[h;h+d]) -------
// Per batch b: M=32, K=64 (tap0: columns u, tap1: columns u+d), N flattened (n,t).
// Block: 256 threads = 8 waves; wave (mh, st) keeps filter+gate accumulators and issues
// 4 chained WMMAs (2 taps x 2 weight sets) sharing the B fragments.
__global__ void gated_wmma_kernel(const u16* __restrict__ hin,
                                  const u16* __restrict__ Wc,   // [4][32 o][32 c] bf16
                                  const float* __restrict__ fb, const float* __restrict__ gb,
                                  u16* __restrict__ hg, int Tout, int d) {
  __shared__ u32 Bf[2][4][32][8];             // 8 KB: tap0 / tap1 tiles, fragment order
  int tid = threadIdx.x;
  int lane = tid & 31, wv = tid >> 5;
  int b = blockIdx.y;
  size_t u0 = (size_t)blockIdx.x * 64;        // < 500*256
  int half = lane >> 4;
  int tcol = lane & 15;
  int mh = wv >> 2;
  int st = wv & 3;
  const size_t slab = (size_t)NP * TSZ;
  const size_t bbase = (size_t)b * CH * slab;
  // ---- stage 2 B tiles (tap0 at u, tap1 at u+d) ----
  int p = tid >> 4;                           // K-pair 0..15 (c = 2p, 2p+1)
  int g = tid & 15;                           // 4-column group
  int lbase = (p >= 8) ? 16 : 0;
  int j = p & 7;
#pragma unroll
  for (int tap = 0; tap < 2; ++tap) {
    const u16* r0 = hin + bbase + (size_t)(2 * p) * slab + u0 + g * 4 + (tap ? d : 0);
    uint2 e = *(const uint2*)r0;
    uint2 f = *(const uint2*)(r0 + slab);
    u32 lo0 = e.x & 0xffffu, lo1 = e.x >> 16, lo2 = e.y & 0xffffu, lo3 = e.y >> 16;
    u32 hi0 = f.x & 0xffffu, hi1 = f.x >> 16, hi2 = f.y & 0xffffu, hi3 = f.y >> 16;
    u32 pk[4] = { lo0 | (hi0 << 16), lo1 | (hi1 << 16), lo2 | (hi2 << 16), lo3 | (hi3 << 16) };
#pragma unroll
    for (int kk = 0; kk < 4; ++kk) {
      int tl = g * 4 + kk;
      Bf[tap][tl >> 4][lbase + (tl & 15)][j] = pk[kk];
    }
  }
  __syncthreads();
  // ---- 4 WMMAs: (tap0,tap1) x (filter, gate), B fragment shared per tap ----
  v8f accf = (v8f){0.f, 0.f, 0.f, 0.f, 0.f, 0.f, 0.f, 0.f};
  v8f accg = accf;
  int arow = mh * 16 + (lane & 15);
#pragma unroll
  for (int tap = 0; tap < 2; ++tap) {
    union { v16bf v; uint4 q[2]; } bf_;
    const u32* bp = &Bf[tap][st][lane][0];
    bf_.q[0] = *(const uint4*)bp;
    bf_.q[1] = *(const uint4*)(bp + 4);
    union { v16bf v; u32 u[8]; } af, ag;
    const u16* wrf = Wc + ((size_t)(tap)     * CH + arow) * CH;
    const u16* wrg = Wc + ((size_t)(2 + tap) * CH + arow) * CH;
#pragma unroll
    for (int jj = 0; jj < 4; ++jj) {
      af.u[jj]     = *(const u32*)(wrf + half * 16 + 2 * jj);
      af.u[4 + jj] = *(const u32*)(wrf + half * 16 + 8 + 2 * jj);
      ag.u[jj]     = *(const u32*)(wrg + half * 16 + 2 * jj);
      ag.u[4 + jj] = *(const u32*)(wrg + half * 16 + 8 + 2 * jj);
    }
    accf = __builtin_amdgcn_wmma_f32_16x16x32_bf16(false, af.v, false, bf_.v,
                                                   (short)0, accf, false, false);
    accg = __builtin_amdgcn_wmma_f32_16x16x32_bf16(false, ag.v, false, bf_.v,
                                                   (short)0, accg, false, false);
  }
  // ---- epilogue: tanh * sigmoid, store bf16 ----
  size_t u = u0 + st * 16 + tcol;
  int t = (int)(u & 255);
  if (t < Tout) {
#pragma unroll
    for (int r = 0; r < 8; ++r) {
      int o = mh * 16 + half * 8 + r;
      float fv = tanhf(accf[r] + fb[o]);
      float gv = 1.f / (1.f + __expf(-(accg[r] + gb[o])));
      hg[bbase + (size_t)o * slab + u] = f2bf(fv * gv);
    }
  }
}

// ---------------- nconv via WMMA bf16: dst[bc,w,t] = sum_v AT[w,v]*src[bc,v,t] ----------------
__global__ void nconv_wmma_kernel(const u16* __restrict__ src, const u16* __restrict__ AT,
                                  u16* __restrict__ dst, int T) {
  __shared__ u32 Bf[4 * 32 * 8];              // [subtile][lane][dword] = 4 KB
  int tid  = threadIdx.x;
  int lane = tid & 31, wv = tid >> 5;
  int bc = blockIdx.z;
  int t0 = blockIdx.y * 64;
  int w0 = blockIdx.x * 64 + wv * 16;
  int half = lane >> 4;
  int mrow = w0 + (lane & 15);
  int tcol = lane & 15;
  const u16* srcbc = src + (size_t)bc * NP * TSZ;
  const u16* arowb = AT + (size_t)mrow * NP;
  v8f acc[4];
#pragma unroll
  for (int s = 0; s < 4; ++s) acc[s] = (v8f){0.f, 0.f, 0.f, 0.f, 0.f, 0.f, 0.f, 0.f};

  for (int ks = 0; ks < 16; ++ks) {
    int v0 = ks * 32;
    __syncthreads();
#pragma unroll
    for (int qq = 0; qq < 2; ++qq) {
      int q = tid + qq * 128;                 // 256 slots: 16 K-pairs x 16 t-groups
      int p = q >> 4;
      int g = q & 15;
      const u16* r0 = srcbc + (size_t)(v0 + 2 * p) * TSZ + (t0 + g * 4);
      uint2 e = *(const uint2*)r0;
      uint2 f = *(const uint2*)(r0 + TSZ);
      if (ks < 15) __builtin_prefetch(r0 + 32 * TSZ, 0, 0);
      u32 lo0 = e.x & 0xffffu, lo1 = e.x >> 16, lo2 = e.y & 0xffffu, lo3 = e.y >> 16;
      u32 hi0 = f.x & 0xffffu, hi1 = f.x >> 16, hi2 = f.y & 0xffffu, hi3 = f.y >> 16;
      int lbase = (p >= 8) ? 16 : 0;
      int j = p & 7;
      u32 pk[4] = { lo0 | (hi0 << 16), lo1 | (hi1 << 16), lo2 | (hi2 << 16), lo3 | (hi3 << 16) };
#pragma unroll
      for (int k = 0; k < 4; ++k) {
        int tl = g * 4 + k;
        int s = tl >> 4, tc = tl & 15;
        Bf[(s * 32 + lbase + tc) * 8 + j] = pk[k];
      }
    }
    __syncthreads();
    union { v16bf v; u32 u[8]; } a;
    const u16* arow = arowb + v0;
    if (ks < 15) __builtin_prefetch(arow + 32, 0, 0);
#pragma unroll
    for (int j = 0; j < 4; ++j) {
      a.u[j]     = *(const u32*)(arow + half * 16 + 2 * j);
      a.u[4 + j] = *(const u32*)(arow + half * 16 + 8 + 2 * j);
    }
#pragma unroll
    for (int s = 0; s < 4; ++s) {
      union { v16bf v; uint4 q[2]; } b;
      const u32* bp = &Bf[(s * 32 + lane) * 8];
      b.q[0] = *(const uint4*)bp;
      b.q[1] = *(const uint4*)(bp + 4);
      acc[s] = __builtin_amdgcn_wmma_f32_16x16x32_bf16(false, a.v, false, b.v,
                                                       (short)0, acc[s], false, false);
    }
  }
  u16* dstbc = dst + (size_t)bc * NP * TSZ;
#pragma unroll
  for (int s = 0; s < 4; ++s) {
    int t = t0 + s * 16 + tcol;
    if (t < T) {
#pragma unroll
      for (int r = 0; r < 8; ++r)
        dstbc[(size_t)(w0 + half * 8 + r) * TSZ + t] = f2bf(acc[s][r]);
    }
  }
}

// ---------------- fused gcn: hout = BN( sum_k W_k*src_k + gbias + residual ) -----------------
__global__ void gcn_fused_kernel(const u16* __restrict__ s0, const u16* __restrict__ s1,
                                 const u16* __restrict__ s2, const u16* __restrict__ s3,
                                 const u16* __restrict__ s4,
                                 const u16* __restrict__ Wb,      // [5][32 o][32 c] bf16
                                 const float* __restrict__ gbias,
                                 const u16* __restrict__ hin, u16* __restrict__ hout,
                                 const float* __restrict__ bng, const float* __restrict__ bnb,
                                 int Tout, int d) {
  __shared__ u32 Bf[5][4][32][8];             // 20 KB, fragment order per source
  int tid = threadIdx.x;
  int lane = tid & 31, wv = tid >> 5;
  int b = blockIdx.y;
  size_t u0 = (size_t)blockIdx.x * 64;        // < 500*256, tiles never touch pad rows
  int half = lane >> 4;
  int tcol = lane & 15;
  int mh = wv >> 2;
  int st = wv & 3;
  const size_t slab = (size_t)NP * TSZ;
  const size_t bbase = (size_t)b * CH * slab;
  int p = tid >> 4;
  int g = tid & 15;
  int lbase = (p >= 8) ? 16 : 0;
  int j = p & 7;
  const u16* srcs[5] = { s0, s1, s2, s3, s4 };
#pragma unroll
  for (int k = 0; k < 5; ++k) {
    const u16* r0 = srcs[k] + bbase + (size_t)(2 * p) * slab + u0 + g * 4;
    uint2 e = *(const uint2*)r0;
    uint2 f = *(const uint2*)(r0 + slab);
    u32 lo0 = e.x & 0xffffu, lo1 = e.x >> 16, lo2 = e.y & 0xffffu, lo3 = e.y >> 16;
    u32 hi0 = f.x & 0xffffu, hi1 = f.x >> 16, hi2 = f.y & 0xffffu, hi3 = f.y >> 16;
    u32 pk[4] = { lo0 | (hi0 << 16), lo1 | (hi1 << 16), lo2 | (hi2 << 16), lo3 | (hi3 << 16) };
#pragma unroll
    for (int kk = 0; kk < 4; ++kk) {
      int tl = g * 4 + kk;
      Bf[k][tl >> 4][lbase + (tl & 15)][j] = pk[kk];
    }
  }
  __syncthreads();
  v8f acc = (v8f){0.f, 0.f, 0.f, 0.f, 0.f, 0.f, 0.f, 0.f};
#pragma unroll
  for (int k = 0; k < 5; ++k) {
    union { v16bf v; u32 u[8]; } a;
    const u16* wr = Wb + ((size_t)k * CH + (mh * 16 + (lane & 15))) * CH;
#pragma unroll
    for (int jj = 0; jj < 4; ++jj) {
      a.u[jj]     = *(const u32*)(wr + half * 16 + 2 * jj);
      a.u[4 + jj] = *(const u32*)(wr + half * 16 + 8 + 2 * jj);
    }
    union { v16bf v; uint4 q[2]; } bf_;
    const u32* bp = &Bf[k][st][lane][0];
    bf_.q[0] = *(const uint4*)bp;
    bf_.q[1] = *(const uint4*)(bp + 4);
    acc = __builtin_amdgcn_wmma_f32_16x16x32_bf16(false, a.v, false, bf_.v,
                                                  (short)0, acc, false, false);
  }
  size_t u = u0 + st * 16 + tcol;
  int t = (int)(u & 255);
  if (t < Tout) {
#pragma unroll
    for (int r = 0; r < 8; ++r) {
      int o = mh * 16 + half * 8 + r;
      size_t idx = bbase + (size_t)o * slab + u;
      float v = acc[r] + gbias[o] + bf2f(hin[idx + d]);
      hout[idx] = f2bf(v * (BN_SCALE * bng[o]) + bnb[o]);
    }
  }
}

// ---------------- skip tap: only the final time column of each layer survives ----------------
__global__ void skip_tap_kernel(const u16* __restrict__ hg, const float* __restrict__ sw,
                                const float* __restrict__ sb, float* __restrict__ skip,
                                int tlast, int init) {
  __shared__ float hv[CH];
  int tid = threadIdx.x;
  int n = blockIdx.x, b = blockIdx.y;
  if (tid < CH) hv[tid] = bf2f(hg[hidx(b, tid, n, tlast)]);
  __syncthreads();
  float v = sb[tid];
#pragma unroll 8
  for (int c = 0; c < CH; ++c) v += sw[tid * 32 + c] * hv[c];
  size_t idx = ((size_t)(b * SC + tid)) * NN + n;
  skip[idx] = init ? v : (skip[idx] + v);
}

// ---------------- fused end MLP: relu(skip) -> relu(512) -> 12 outputs -----------------------
__global__ void end_kernel(const float* __restrict__ skip, const float* __restrict__ w1,
                           const float* __restrict__ b1, const float* __restrict__ w2,
                           const float* __restrict__ b2, float* __restrict__ out) {
  __shared__ float sk[SC];
  __shared__ float h1[EC];
  __shared__ float red[192];
  int tid = threadIdx.x;
  int n = blockIdx.x, b = blockIdx.y;
  sk[tid] = fmaxf(skip[((size_t)(b * SC + tid)) * NN + n], 0.f);
  __syncthreads();
  for (int e = tid; e < EC; e += 256) {
    float v = b1[e];
    const float* wr = w1 + (size_t)e * SC;
#pragma unroll 8
    for (int s = 0; s < SC; ++s) v += wr[s] * sk[s];
    h1[e] = fmaxf(v, 0.f);
  }
  __syncthreads();
  if (tid < 192) {
    int o = tid >> 4, seg = tid & 15;
    float p = 0.f;
    const float* wr = w2 + (size_t)o * EC + seg * 32;
#pragma unroll
    for (int k = 0; k < 32; ++k) p += wr[k] * h1[seg * 32 + k];
    red[tid] = p;
  }
  __syncthreads();
  if (tid < OD) {
    float v = b2[tid];
#pragma unroll
    for (int g = 0; g < 16; ++g) v += red[tid * 16 + g];
    out[((size_t)(b * OD + tid)) * NN + n] = v;
  }
}

extern "C" void kernel_launch(void* const* d_in, const int* in_sizes, int n_in,
                              void* d_out, int out_size, void* d_ws, size_t ws_size,
                              hipStream_t stream) {
  const float* x       = (const float*)d_in[0];
  const float* A0      = (const float*)d_in[1];
  const float* start_w = (const float*)d_in[2];
  const float* start_b = (const float*)d_in[3];
  const float* filt_w  = (const float*)d_in[4];
  const float* filt_b  = (const float*)d_in[5];
  const float* gate_w  = (const float*)d_in[6];
  const float* gate_b  = (const float*)d_in[7];
  const float* skip_w  = (const float*)d_in[8];
  const float* skip_b  = (const float*)d_in[9];
  const float* gcn_w   = (const float*)d_in[10];
  const float* gcn_b   = (const float*)d_in[11];
  const float* bn_g    = (const float*)d_in[12];
  const float* bn_b    = (const float*)d_in[13];
  const float* nv1     = (const float*)d_in[14];
  const float* nv2     = (const float*)d_in[15];
  const float* e1w     = (const float*)d_in[16];
  const float* e1b     = (const float*)d_in[17];
  const float* e2w     = (const float*)d_in[18];
  const float* e2b     = (const float*)d_in[19];
  float* out = (float*)d_out;

  char* ws = (char*)d_ws;
  const size_t szAT  = (size_t)NP * NP * 2;
  const size_t szH   = (size_t)B_ * CH * NP * TSZ * 2;
  const size_t szWb  = 73728;   // (LL-1)*5*32*32*2 = 71680, rounded
  const size_t szWc  = 65536;   // LL*4*32*32*2
  u16*   ATp   = (u16*)(ws);
  u16*   ADPTp = (u16*)(ws + szAT);
  u16*   hA    = (u16*)(ws + 2 * szAT);
  u16*   hB    = (u16*)(ws + 2 * szAT + 1 * szH);
  u16*   hg    = (u16*)(ws + 2 * szAT + 2 * szH);
  u16*   t1    = (u16*)(ws + 2 * szAT + 3 * szH);
  u16*   t2    = (u16*)(ws + 2 * szAT + 4 * szH);
  u16*   t3    = (u16*)(ws + 2 * szAT + 5 * szH);
  u16*   t4    = (u16*)(ws + 2 * szAT + 6 * szH);
  u16*   Wb    = (u16*)(ws + 2 * szAT + 7 * szH);
  u16*   Wc    = (u16*)(ws + 2 * szAT + 7 * szH + szWb);
  float* skp   = (float*)(ws + 2 * szAT + 7 * szH + szWb + szWc);

  transpose_pad_kernel<<<(NP * NP) / 256, 256, 0, stream>>>(A0, ATp, ADPTp);
  adp_softmax_kernel<<<NN, 256, 0, stream>>>(nv1, nv2, ADPTp);
  wb_convert_kernel<<<((LL - 1) * 5 * CH * CH + 255) / 256, 256, 0, stream>>>(gcn_w, Wb);
  wc_convert_kernel<<<(LL * 4 * CH * CH + 255) / 256, 256, 0, stream>>>(filt_w, gate_w, Wc);
  start_conv_kernel<<<(B_ * CH * NN * TSZ) / 256, 256, 0, stream>>>(x, start_w, start_b, hA);

  u16* hin = hA; u16* hout = hB;
  int T = TSZ;
  for (int i = 0; i < LL; ++i) {
    int d = 1 << i;
    int Tout = T - d;
    gated_wmma_kernel<<<dim3((NN * TSZ) / 64, B_), 256, 0, stream>>>(
        hin, Wc + (size_t)i * 4 * CH * CH, filt_b + i * 32, gate_b + i * 32, hg, Tout, d);
    skip_tap_kernel<<<dim3(NN, B_), 256, 0, stream>>>(
        hg, skip_w + (size_t)i * SC * CH, skip_b + i * SC, skp, Tout - 1, (i == 0) ? 1 : 0);
    if (i < LL - 1) {  // last layer's h is dead code: only skip survives
      dim3 gn(8, (Tout + 63) / 64, B_ * CH);
      nconv_wmma_kernel<<<gn, 128, 0, stream>>>(hg, ATp,   t1, Tout);
      nconv_wmma_kernel<<<gn, 128, 0, stream>>>(t1, ATp,   t2, Tout);
      nconv_wmma_kernel<<<gn, 128, 0, stream>>>(hg, ADPTp, t3, Tout);
      nconv_wmma_kernel<<<gn, 128, 0, stream>>>(t3, ADPTp, t4, Tout);
      gcn_fused_kernel<<<dim3((NN * TSZ) / 64, B_), 256, 0, stream>>>(
          hg, t1, t2, t3, t4, Wb + (size_t)i * 5 * CH * CH, gcn_b + i * 32,
          hin, hout, bn_g + i * 32, bn_b + i * 32, Tout, d);
      u16* tmp = hin; hin = hout; hout = tmp;
    }
    T = Tout;
  }
  end_kernel<<<dim3(NN, B_), 256, 0, stream>>>(skp, e1w, e1b, e2w, e2b, out);
  (void)in_sizes; (void)n_in; (void)out_size; (void)ws_size;
}
